// TRTViTBlock_74964359184709
// MI455X (gfx1250) — compile-verified, matrix operand order checked by reading, not run on hardware
//
#include <hip/hip_runtime.h>
#include <hip/hip_bf16.h>
#include <math.h>

// ---------------- problem constants ----------------
#define DIMC   768
#define NHEAD  12
#define HDIM   64
#define HSZ    32
#define WSZ    32
#define BSZ    4
#define LSEQ   1024          // 32*32
#define HIDC   3072
#define NBH    (BSZ * NHEAD) // 48 (b,head) pairs
#define NTOK   (BSZ * LSEQ)  // 4096 rows

typedef __bf16 bf16;
typedef __attribute__((ext_vector_type(16))) __bf16 v16bf;
typedef __attribute__((ext_vector_type(8)))  float  v8f;

union Frag {
  v16bf        v;
  unsigned int u[8];
  __bf16       e[16];
};

static __device__ __forceinline__ v8f wmma_bf16(v16bf a, v16bf b, v8f c) {
  // D = A(16x32) * B(32x16) + C, fp32 accumulate
  return __builtin_amdgcn_wmma_f32_16x16x32_bf16(false, a, false, b, (short)0, c,
                                                 false, false);
}

// K-offset pattern for 16-bit A/B fragments (pairs per VGPR), per ISA 7.12.2
static __device__ __forceinline__ int kpat(int i) {
  return (i < 4) ? (2 * i) : (16 + 2 * (i - 4));
}

// ---- CDNA5 async global->LDS copy (no VGPR data path, ASYNCcnt-tracked) ----
// LDS aperture addresses carry the byte offset in addr[31:0] (ISA 10.2), so
// truncating the generic shared-pointer address yields the VDST LDS address.
static __device__ __forceinline__ void async_b128(void* lds, const void* g) {
  asm volatile("global_load_async_to_lds_b128 %0, %1, off"
               :: "v"((unsigned)(size_t)lds),
                  "v"((unsigned long long)(size_t)g)
               : "memory");
}
static __device__ __forceinline__ void wait_async0() {
  asm volatile("s_wait_asynccnt 0x0" ::: "memory");
}

// ======================================================================
// LayerNorm over 768, fp32 in -> bf16 out.  One 256-thread block per row.
// ======================================================================
__global__ __launch_bounds__(256) void ln_kernel(const float* __restrict__ x,
                                                 const float* __restrict__ g,
                                                 const float* __restrict__ b,
                                                 bf16* __restrict__ out) {
  __shared__ float red[256];
  const int row = blockIdx.x;
  const int tid = threadIdx.x;
  const float* xr = x + (size_t)row * DIMC;

  float lv[3];
#pragma unroll
  for (int i = 0; i < 3; ++i) lv[i] = xr[tid + i * 256];

  float s = lv[0] + lv[1] + lv[2];
  red[tid] = s;
  __syncthreads();
  for (int off = 128; off > 0; off >>= 1) {
    if (tid < off) red[tid] += red[tid + off];
    __syncthreads();
  }
  const float mu = red[0] * (1.0f / DIMC);
  __syncthreads();

  float d0 = lv[0] - mu, d1 = lv[1] - mu, d2 = lv[2] - mu;
  red[tid] = d0 * d0 + d1 * d1 + d2 * d2;
  __syncthreads();
  for (int off = 128; off > 0; off >>= 1) {
    if (tid < off) red[tid] += red[tid + off];
    __syncthreads();
  }
  const float rs = rsqrtf(red[0] * (1.0f / DIMC) + 1e-5f);

#pragma unroll
  for (int i = 0; i < 3; ++i) {
    const int c = tid + i * 256;
    out[(size_t)row * DIMC + c] = (bf16)((lv[i] - mu) * rs * g[c] + b[c]);
  }
}

// ======================================================================
// Weight convert: fp32 -> bf16, SAME (N,K) row-major layout.
// ======================================================================
__global__ __launch_bounds__(256) void cvt_kernel(const float* __restrict__ src,
                                                  bf16* __restrict__ dst,
                                                  int count4) {
  const int t = blockIdx.x * 256 + threadIdx.x;
  if (t >= count4) return;
  const float4 v = *(const float4*)&src[(size_t)t * 4];
  bf16* d = dst + (size_t)t * 4;
  d[0] = (bf16)v.x; d[1] = (bf16)v.y; d[2] = (bf16)v.z; d[3] = (bf16)v.w;
}

// ======================================================================
// Generic bf16 WMMA GEMM: C(MxN) = A(MxK) * Bw(N,K)^T + bias (+epilogue)
// Block tile 64x256, 8 waves of 32x64, K-step 32, double-buffered LDS filled
// with async global->LDS b128 copies (ASYNCcnt).
// mode 0: fp32 out = v ; mode 1: bf16 out = gelu(v) ; mode 2: fp32 out = v+res
// ======================================================================
__global__ __launch_bounds__(256) void gemm_bf16_kernel(
    const bf16* __restrict__ A, const bf16* __restrict__ Bw,
    const float* __restrict__ bias, const float* __restrict__ res,
    void* __restrict__ out, int M, int N, int K, int mode) {
  __shared__ bf16 As[2][64][40];    // rows padded to 80B (16B multiple)
  __shared__ bf16 Bs[2][256][40];   // N-major: Bs[.][n][k]

  const int tid = threadIdx.x;
  const int lane = tid & 31, wv = tid >> 5;
  const int wm = wv >> 2, wn = wv & 3;
  const int ln16 = lane & 15, laneHi = lane >> 4;
  const int mtile = blockIdx.y * 64;
  const int ntile = blockIdx.x * 256;

  // per-thread b128 tile-copy coordinates
  const int ar_ = tid >> 2;        // A row 0..63
  const int ac_ = (tid & 3) * 8;   // A col 0,8,16,24

  auto issueTile = [&](int k0, int bsel) {
    async_b128(&As[bsel][ar_][ac_], &A[(size_t)(mtile + ar_) * K + k0 + ac_]);
#pragma unroll
    for (int i = 0; i < 4; ++i) {
      const int idx = tid + i * 256;
      async_b128(&Bs[bsel][idx >> 2][(idx & 3) * 8],
                 &Bw[(size_t)(ntile + (idx >> 2)) * K + k0 + (idx & 3) * 8]);
    }
  };

  v8f acc[2][4];
#pragma unroll
  for (int f = 0; f < 2; ++f)
#pragma unroll
    for (int s = 0; s < 4; ++s) acc[f][s] = (v8f)0.0f;

  issueTile(0, 0);
  int buf = 0;

  for (int k0 = 0; k0 < K; k0 += 32) {
    wait_async0();     // my async writes into buf are complete
    __syncthreads();   // everyone's writes complete -> tiles[buf] ready
    if (k0 + 32 < K) issueTile(k0 + 32, buf ^ 1);  // next tile in flight

    Frag a[2];
#pragma unroll
    for (int f = 0; f < 2; ++f) {
      const int m = wm * 32 + f * 16 + ln16;
#pragma unroll
      for (int i = 0; i < 8; ++i) {
        const int kb = kpat(i) + 8 * laneHi;
        a[f].u[i] = *(const unsigned int*)&As[buf][m][kb];
      }
    }
#pragma unroll
    for (int s = 0; s < 4; ++s) {
      Frag bf;
      const int n = wn * 64 + s * 16 + ln16;
#pragma unroll
      for (int i = 0; i < 8; ++i) {
        const int kb = kpat(i) + 8 * laneHi;
        bf.u[i] = *(const unsigned int*)&Bs[buf][n][kb];
      }
      acc[0][s] = wmma_bf16(a[0].v, bf.v, acc[0][s]);
      acc[1][s] = wmma_bf16(a[1].v, bf.v, acc[1][s]);
    }

    buf ^= 1;
  }

#pragma unroll
  for (int f = 0; f < 2; ++f)
#pragma unroll
    for (int s = 0; s < 4; ++s)
#pragma unroll
      for (int r = 0; r < 8; ++r) {
        const int m = mtile + wm * 32 + f * 16 + r + 8 * laneHi;
        const int n = ntile + wn * 64 + s * 16 + ln16;
        float v = acc[f][s][r] + bias[n];
        if (mode == 2) {
          ((float*)out)[(size_t)m * N + n] = v + res[(size_t)m * N + n];
        } else if (mode == 1) {
          v = 0.5f * v * (1.0f + erff(v * 0.70710678118f));  // exact GELU
          ((bf16*)out)[(size_t)m * N + n] = (bf16)v;
        } else {
          ((float*)out)[(size_t)m * N + n] = v;
        }
      }
}

// ======================================================================
// RoPE + repack: qkv fp32 (NTOK, 2304) -> Q,K (rotated) bf16 (NBH,LSEQ,HDIM),
// V bf16 head-transposed (NBH, HDIM, LSEQ).  1 thread per (bn,l,pair).
// ======================================================================
__global__ __launch_bounds__(256) void rope_kernel(const float* __restrict__ qkv,
                                                   bf16* __restrict__ Q,
                                                   bf16* __restrict__ Kp,
                                                   bf16* __restrict__ V) {
  const int t = blockIdx.x * 256 + threadIdx.x;
  if (t >= NBH * LSEQ * 32) return;
  const int j  = t & 31;
  const int l  = (t >> 5) & (LSEQ - 1);
  const int bn = t >> 15;
  const int b = bn / NHEAD, nh = bn % NHEAD;

  const float* base = qkv + (size_t)(b * LSEQ + l) * (3 * DIMC) + nh * HDIM;
  const int jj = j & 15;
  const float freq = __powf(10000.0f, -(4.0f * (float)jj) / 64.0f);
  const float pos = (j < 16) ? (float)(l & 31) : (float)(l >> 5);
  float sn, cs;
  __sincosf(pos * freq, &sn, &cs);

  const size_t o = (size_t)(bn * LSEQ + l) * HDIM + 2 * j;
  const float q0 = base[2 * j], q1 = base[2 * j + 1];
  Q[o]     = (bf16)(q0 * cs - q1 * sn);
  Q[o + 1] = (bf16)(q0 * sn + q1 * cs);
  const float k0 = base[DIMC + 2 * j], k1 = base[DIMC + 2 * j + 1];
  Kp[o]     = (bf16)(k0 * cs - k1 * sn);
  Kp[o + 1] = (bf16)(k0 * sn + k1 * cs);
  // V transposed per head: V[bn][c][l]
  V[((size_t)bn * HDIM + 2 * j) * LSEQ + l]     = (bf16)base[2 * DIMC + 2 * j];
  V[((size_t)bn * HDIM + 2 * j + 1) * LSEQ + l] = (bf16)base[2 * DIMC + 2 * j + 1];
}

// ======================================================================
// Decomposed rel-pos tables: relH[bn,q,kh] = <Q[bn,q,:], rel_pos_h[h-kh+31,:]>
// and relW likewise with w-kw+31.  1 thread per (bn,q,j); dot of 64.
// ======================================================================
__global__ __launch_bounds__(256) void relpos_kernel(const bf16* __restrict__ Q,
                                                     const float* __restrict__ rph,
                                                     const float* __restrict__ rpw,
                                                     float* __restrict__ relH,
                                                     float* __restrict__ relW) {
  const int t = blockIdx.x * 256 + threadIdx.x;
  if (t >= NBH * LSEQ * 32) return;
  const int j  = t & 31;
  const int q  = (t >> 5) & (LSEQ - 1);
  const int bn = t >> 15;
  const int h = q >> 5, w = q & 31;

  const bf16* qv = Q + (size_t)(bn * LSEQ + q) * HDIM;
  const float* ph = rph + (size_t)(h - j + HSZ - 1) * HDIM;
  const float* pw = rpw + (size_t)(w - j + WSZ - 1) * HDIM;
  float sh = 0.f, sw = 0.f;
#pragma unroll 8
  for (int c = 0; c < HDIM; ++c) {
    const float qq = (float)qv[c];
    sh += qq * ph[c];
    sw += qq * pw[c];
  }
  relH[t] = sh;
  relW[t] = sw;
}

// ======================================================================
// Attention, flash-style online softmax.  Grid = NBH * (LSEQ/128).
// 8 waves; wave owns 16 query rows; key tiles of 64, double-buffered LDS
// filled with async global->LDS b128 copies.
// scores = 0.125*Q K^T + relH + relW ; out (bf16) -> (NTOK, DIMC) layout.
// ======================================================================
__global__ __launch_bounds__(256) void attn_kernel(const bf16* __restrict__ Q,
                                                   const bf16* __restrict__ Kg,
                                                   const bf16* __restrict__ Vg,
                                                   const float* __restrict__ relH,
                                                   const float* __restrict__ relW,
                                                   bf16* __restrict__ out) {
  __shared__ bf16 Kt[2][64][72];    // (key, c), rows padded to 144B
  __shared__ bf16 Vt[2][64][72];    // (c, key) c-major so pairs are contiguous
  __shared__ bf16 Pl[8][16][72];    // per-wave P tiles (m, key)
  __shared__ float rh[128][32];
  __shared__ float rw[128][32];

  const int bn = blockIdx.x >> 3;
  const int qblk = blockIdx.x & 7;
  const int b = bn / NHEAD, nh = bn % NHEAD;
  const int tid = threadIdx.x;
  const int lane = tid & 31, wv = tid >> 5;
  const int ln16 = lane & 15, laneHi = lane >> 4;
  const int qwg = qblk * 128;

  const bf16* Kbase = Kg + (size_t)bn * LSEQ * HDIM;   // (key, c)
  const bf16* Vbase = Vg + (size_t)bn * HDIM * LSEQ;   // (c, key)

  auto issueTiles = [&](int kt, int bsel) {
#pragma unroll
    for (int i = 0; i < 2; ++i) {
      const int idx = tid + i * 256;
      const int r = idx >> 3, c = (idx & 7) * 8;
      async_b128(&Kt[bsel][r][c], &Kbase[(size_t)(kt * 64 + r) * HDIM + c]);
      async_b128(&Vt[bsel][r][c], &Vbase[(size_t)r * LSEQ + kt * 64 + c]);
    }
  };

  // stage rel-pos rows for this query block
  {
    const float* pH = relH + (size_t)(bn * LSEQ + qwg) * 32;
    const float* pW = relW + (size_t)(bn * LSEQ + qwg) * 32;
    for (int i = tid; i < 128 * 32; i += 256) {
      rh[i >> 5][i & 31] = pH[i];
      rw[i >> 5][i & 31] = pW[i];
    }
  }

  // Q fragments for this wave's 16 rows (K = 64 -> 2 A-frags), kept in regs
  Frag qa[2];
  {
    const bf16* Qb = Q + (size_t)(bn * LSEQ + qwg + wv * 16) * HDIM;
#pragma unroll
    for (int f = 0; f < 2; ++f)
#pragma unroll
      for (int i = 0; i < 8; ++i) {
        const int kb = kpat(i) + 8 * laneHi + f * 32;
        qa[f].u[i] = *(const unsigned int*)&Qb[(size_t)ln16 * HDIM + kb];
      }
  }

  float Mr[8], Lr[8];
  v8f O[4];
#pragma unroll
  for (int r = 0; r < 8; ++r) { Mr[r] = -1e30f; Lr[r] = 0.f; }
#pragma unroll
  for (int s = 0; s < 4; ++s) O[s] = (v8f)0.0f;

  issueTiles(0, 0);
  int buf = 0;

  for (int kt = 0; kt < LSEQ / 64; ++kt) {
    wait_async0();     // async writes into tiles[buf] complete (this thread)
    __syncthreads();   // all threads' writes complete; prev Pl consumed
    const bool nxt = (kt + 1) < (LSEQ / 64);
    if (nxt) issueTiles(kt + 1, buf ^ 1);  // next K/V tiles in flight

    // ---- scores S = 0.125 * Q K^T + bias ----
    float S[4][8];
#pragma unroll
    for (int s = 0; s < 4; ++s) {
      Frag kb0, kb1;
      const int key = s * 16 + ln16;
#pragma unroll
      for (int i = 0; i < 8; ++i) {
        const int kb = kpat(i) + 8 * laneHi;
        kb0.u[i] = *(const unsigned int*)&Kt[buf][key][kb];
        kb1.u[i] = *(const unsigned int*)&Kt[buf][key][kb + 32];
      }
      v8f sc = wmma_bf16(qa[0].v, kb0.v, (v8f)0.0f);
      sc = wmma_bf16(qa[1].v, kb1.v, sc);
#pragma unroll
      for (int r = 0; r < 8; ++r) {
        const int m = r + 8 * laneHi;
        const int gkey = kt * 64 + s * 16 + ln16;
        S[s][r] = sc[r] * 0.125f + rh[wv * 16 + m][gkey >> 5] +
                  rw[wv * 16 + m][gkey & 31];
      }
    }

    // ---- online softmax ----
    float tmax[8];
#pragma unroll
    for (int r = 0; r < 8; ++r)
      tmax[r] = fmaxf(fmaxf(S[0][r], S[1][r]), fmaxf(S[2][r], S[3][r]));
#pragma unroll
    for (int off = 8; off > 0; off >>= 1)
#pragma unroll
      for (int r = 0; r < 8; ++r)
        tmax[r] = fmaxf(tmax[r], __shfl_xor(tmax[r], off, 32));

    float al[8], rsum[8];
#pragma unroll
    for (int r = 0; r < 8; ++r) {
      const float Mn = fmaxf(Mr[r], tmax[r]);
      al[r] = __expf(Mr[r] - Mn);
      Mr[r] = Mn;
      rsum[r] = 0.f;
    }
#pragma unroll
    for (int s = 0; s < 4; ++s)
#pragma unroll
      for (int r = 0; r < 8; ++r) {
        const float p = __expf(S[s][r] - Mr[r]);
        S[s][r] = p;
        rsum[r] += p;
      }
#pragma unroll
    for (int off = 8; off > 0; off >>= 1)
#pragma unroll
      for (int r = 0; r < 8; ++r) rsum[r] += __shfl_xor(rsum[r], off, 32);
#pragma unroll
    for (int r = 0; r < 8; ++r) Lr[r] = Lr[r] * al[r] + rsum[r];
#pragma unroll
    for (int s = 0; s < 4; ++s)
#pragma unroll
      for (int r = 0; r < 8; ++r) O[s][r] *= al[r];

    // ---- transpose P through LDS into A-fragment layout ----
#pragma unroll
    for (int s = 0; s < 4; ++s)
#pragma unroll
      for (int r = 0; r < 8; ++r)
        Pl[wv][r + 8 * laneHi][s * 16 + ln16] = (bf16)S[s][r];
    __syncthreads();  // P tiles ready

    Frag pa[2];
#pragma unroll
    for (int f = 0; f < 2; ++f)
#pragma unroll
      for (int i = 0; i < 8; ++i) {
        const int kb = kpat(i) + 8 * laneHi + f * 32;
        pa[f].u[i] = *(const unsigned int*)&Pl[wv][ln16][kb];
      }

    // ---- O += P V : V fragments from c-major Vt, pairs contiguous ----
#pragma unroll
    for (int cs = 0; cs < 4; ++cs) {
      Frag vb0, vb1;
      const int n = cs * 16 + ln16;  // output channel c
#pragma unroll
      for (int i = 0; i < 8; ++i) {
        const int kb = kpat(i) + 8 * laneHi;
        vb0.u[i] = *(const unsigned int*)&Vt[buf][n][kb];
        vb1.u[i] = *(const unsigned int*)&Vt[buf][n][kb + 32];
      }
      O[cs] = wmma_bf16(pa[0].v, vb0.v, O[cs]);
      O[cs] = wmma_bf16(pa[1].v, vb1.v, O[cs]);
    }

    buf ^= 1;
  }

  // ---- normalize and store bf16 into (NTOK, DIMC) layout ----
  bf16* ob = out + (size_t)(b * LSEQ + qwg + wv * 16) * DIMC + nh * HDIM;
#pragma unroll
  for (int cs = 0; cs < 4; ++cs)
#pragma unroll
    for (int r = 0; r < 8; ++r) {
      const int m = r + 8 * laneHi;
      ob[(size_t)m * DIMC + cs * 16 + ln16] = (bf16)(O[cs][r] / Lr[r]);
    }
}

// ======================================================================
// Host-side launcher.  Workspace layout ~134 MiB (assumed available).
// ======================================================================
extern "C" void kernel_launch(void* const* d_in, const int* in_sizes, int n_in,
                              void* d_out, int out_size, void* d_ws, size_t ws_size,
                              hipStream_t stream) {
  (void)in_sizes; (void)n_in; (void)out_size; (void)ws_size;
  const float* x      = (const float*)d_in[0];
  const float* ln1_w  = (const float*)d_in[1];
  const float* ln1_b  = (const float*)d_in[2];
  const float* w_qkv  = (const float*)d_in[3];
  const float* b_qkv  = (const float*)d_in[4];
  const float* w_proj = (const float*)d_in[5];
  const float* b_proj = (const float*)d_in[6];
  const float* rph    = (const float*)d_in[7];
  const float* rpw    = (const float*)d_in[8];
  const float* ln2_w  = (const float*)d_in[9];
  const float* ln2_b  = (const float*)d_in[10];
  const float* w_fc1  = (const float*)d_in[11];
  const float* b_fc1  = (const float*)d_in[12];
  const float* w_fc2  = (const float*)d_in[13];
  const float* b_fc2  = (const float*)d_in[14];

  char* ws = (char*)d_ws;
  size_t off = 0;
  auto alloc = [&](size_t bytes) -> void* {
    void* p = ws + off;
    off = (off + bytes + 255) & ~(size_t)255;
    return p;
  };
  bf16*  qkvW  = (bf16*) alloc((size_t)3 * DIMC * DIMC * 2);   // (2304,768)
  bf16*  projW = (bf16*) alloc((size_t)DIMC * DIMC * 2);       // (768,768)
  bf16*  fc1W  = (bf16*) alloc((size_t)HIDC * DIMC * 2);       // (3072,768)
  bf16*  fc2W  = (bf16*) alloc((size_t)DIMC * HIDC * 2);       // (768,3072)
  bf16*  xn    = (bf16*) alloc((size_t)NTOK * DIMC * 2);
  float* qkv   = (float*)alloc((size_t)NTOK * 3 * DIMC * 4);
  bf16*  Qb    = (bf16*) alloc((size_t)NBH * LSEQ * HDIM * 2);
  bf16*  Kb    = (bf16*) alloc((size_t)NBH * LSEQ * HDIM * 2);
  bf16*  Vb    = (bf16*) alloc((size_t)NBH * LSEQ * HDIM * 2);
  float* relH  = (float*)alloc((size_t)NBH * LSEQ * 32 * 4);
  float* relW  = (float*)alloc((size_t)NBH * LSEQ * 32 * 4);
  bf16*  aout  = (bf16*) alloc((size_t)NTOK * DIMC * 2);
  float* X1    = (float*)alloc((size_t)NTOK * DIMC * 4);
  bf16*  hbuf  = (bf16*) alloc((size_t)NTOK * DIMC * 2);
  bf16*  hact  = (bf16*) alloc((size_t)NTOK * HIDC * 2);

  // 1) weight converts (keep (N,K) layout)
  cvt_kernel<<<(3 * DIMC * DIMC / 4 + 255) / 256, 256, 0, stream>>>(w_qkv,  qkvW,  3 * DIMC * DIMC / 4);
  cvt_kernel<<<(DIMC * DIMC / 4 + 255) / 256,     256, 0, stream>>>(w_proj, projW, DIMC * DIMC / 4);
  cvt_kernel<<<(HIDC * DIMC / 4 + 255) / 256,     256, 0, stream>>>(w_fc1,  fc1W,  HIDC * DIMC / 4);
  cvt_kernel<<<(DIMC * HIDC / 4 + 255) / 256,     256, 0, stream>>>(w_fc2,  fc2W,  DIMC * HIDC / 4);

  // 2) LN1
  ln_kernel<<<NTOK, 256, 0, stream>>>(x, ln1_w, ln1_b, xn);

  // 3) QKV GEMM (4096x768 @ 768x2304) -> fp32
  gemm_bf16_kernel<<<dim3(3 * DIMC / 256, NTOK / 64), 256, 0, stream>>>(
      xn, qkvW, b_qkv, nullptr, qkv, NTOK, 3 * DIMC, DIMC, 0);

  // 4) RoPE + per-head repack (V transposed per head)
  rope_kernel<<<(NBH * LSEQ * 32) / 256, 256, 0, stream>>>(qkv, Qb, Kb, Vb);

  // 5) rel-pos tables
  relpos_kernel<<<(NBH * LSEQ * 32) / 256, 256, 0, stream>>>(Qb, rph, rpw, relH, relW);

  // 6) attention
  attn_kernel<<<NBH * (LSEQ / 128), 256, 0, stream>>>(Qb, Kb, Vb, relH, relW, aout);

  // 7) proj GEMM + residual x -> X1 (fp32)
  gemm_bf16_kernel<<<dim3(DIMC / 256, NTOK / 64), 256, 0, stream>>>(
      aout, projW, b_proj, x, X1, NTOK, DIMC, DIMC, 2);

  // 8) LN2
  ln_kernel<<<NTOK, 256, 0, stream>>>(X1, ln2_w, ln2_b, hbuf);

  // 9) FC1 + exact GELU -> bf16
  gemm_bf16_kernel<<<dim3(HIDC / 256, NTOK / 64), 256, 0, stream>>>(
      hbuf, fc1W, b_fc1, nullptr, hact, NTOK, HIDC, DIMC, 1);

  // 10) FC2 + residual X1 -> d_out (fp32)
  gemm_bf16_kernel<<<dim3(DIMC / 256, NTOK / 64), 256, 0, stream>>>(
      hact, fc2W, b_fc2, X1, d_out, NTOK, DIMC, HIDC, 2);
}
